// Mamba2Layer_28063316312548
// MI455X (gfx1250) — matrix operand order repaired and checked by reading, not compiled, not run on hardware
//
#include <hip/hip_runtime.h>

// ---------------------------------------------------------------------------
// Mamba2 layer for MI455X (gfx1250, wave32, WMMA).
//   H=16, CH=64, N=64, D=1024, tokens = B*T = 16384.
// bf16 operands + f32 WMMA accumulation for all GEMMs, including the
// chunked (Q=64) state-space scan; log-space decay handling for stability.
// Output GEMM uses double-buffered GLOBAL_LOAD_ASYNC_TO_LDS staging.
// ---------------------------------------------------------------------------

#define H_    16
#define CH_   64
#define NS_   64
#define D_    1024
#define TSEQ  4096
#define TTOK  16384   // B*T
#define Q_    64      // scan chunk length
#define STR_  72      // LDS row stride (halfwords / floats): 144B, 16B-aligned frags

#if __has_builtin(__builtin_amdgcn_global_load_async_to_lds_b128)
#define HAVE_ASYNC_LDS 1
#else
#define HAVE_ASYNC_LDS 0
#endif

typedef __attribute__((ext_vector_type(16))) __bf16 v16bf;
typedef __attribute__((ext_vector_type(8)))  __bf16 v8bf;
typedef __attribute__((ext_vector_type(8)))  float  v8f;
typedef int vi4 __attribute__((vector_size(16)));    // matches builtin's V4i

struct u4x2 { uint4 a, b; };   // 32B aggregate for whole-fragment bit_cast

__device__ __forceinline__ unsigned short f2bfu(float f) {
    unsigned u = __builtin_bit_cast(unsigned, f);
    u += 0x7FFFu + ((u >> 16) & 1u);
    return (unsigned short)(u >> 16);
}
__device__ __forceinline__ float bf2f(unsigned short u) {
    return __builtin_bit_cast(float, ((unsigned)u) << 16);
}

__device__ __forceinline__ v8f bfmma(v16bf a, v16bf b, v8f c) {
    return __builtin_amdgcn_wmma_f32_16x16x32_bf16(false, a, false, b, (short)0, c, false, false);
}

// --- WMMA fragment loaders (CDNA5 ISA 7.12.2: 16-bit 16x32 A / 32x16 B) ---

// A fragment from fp32 row (convert on the fly); elements 0..7 = K base..base+7,
// elements 8..15 = K base+16..base+23 (base = kb + 8*half).
__device__ __forceinline__ v16bf packA_f32(const float* __restrict__ xrow, int base) {
    v16bf a;
#pragma unroll
    for (int i = 0; i < 8; ++i)
        a[i] = __builtin_bit_cast(__bf16, f2bfu(xrow[base + i]));
#pragma unroll
    for (int i = 0; i < 8; ++i)
        a[8 + i] = __builtin_bit_cast(__bf16, f2bfu(xrow[base + 16 + i]));
    return a;
}

// A fragment from bf16 memory (global or LDS; two 16B loads, single bit_cast).
__device__ __forceinline__ v16bf loadA_bf16(const unsigned short* __restrict__ row, int base) {
    u4x2 t;
    t.a = *(const uint4*)(row + base);
    t.b = *(const uint4*)(row + base + 16);
    return __builtin_bit_cast(v16bf, t);
}

// B fragment: lane holds W[n, krun .. krun+7] and W[n, krun+16 .. krun+23].
__device__ __forceinline__ v16bf loadB_bf16(const unsigned short* __restrict__ wrow) {
    u4x2 t;
    t.a = *(const uint4*)(wrow);
    t.b = *(const uint4*)(wrow + 16);
    return __builtin_bit_cast(v16bf, t);
}

// LDS helpers (stride STR_). After inlining, clang recovers addrspace(3) -> ds ops.
__device__ __forceinline__ v16bf ldsA(const unsigned short* sm, int row, int kb, int hf) {
    return loadA_bf16(sm + row * STR_, kb + hf * 8);
}
__device__ __forceinline__ v16bf ldsB(const unsigned short* sm, int row, int kb, int hf) {
    return loadB_bf16(sm + row * STR_ + kb + hf * 8);
}
// B fragment from f32 LDS (state), converted to bf16.
__device__ __forceinline__ v16bf ldsB_f32(const float* sm, int row, int kb, int hf) {
    const float* p = sm + row * STR_ + kb + hf * 8;
    v16bf b;
#pragma unroll
    for (int i = 0; i < 16; ++i)
        b[i] = __builtin_bit_cast(__bf16, f2bfu(p[i + (i & 8)]));
    return b;
}

// ---- async global->LDS copy (ASYNCcnt path), sync fallback ----
__device__ __forceinline__ void async_cp_b128(const unsigned short* g, unsigned short* l) {
#if HAVE_ASYNC_LDS
    __builtin_amdgcn_global_load_async_to_lds_b128(
        (__attribute__((address_space(1))) vi4*)(unsigned long long)g,
        (__attribute__((address_space(3))) vi4*)l,
        0, 0);
#else
    *(uint4*)l = *(const uint4*)g;
#endif
}
__device__ __forceinline__ void wait_async_all() {
#if HAVE_ASYNC_LDS
#if __has_builtin(__builtin_amdgcn_s_wait_asynccnt)
    __builtin_amdgcn_s_wait_asynccnt(0);
#else
    asm volatile("s_wait_asynccnt 0" ::: "memory");
#endif
#endif
}

// ---------------------------------------------------------------------------
// K0: fp32 -> bf16 weight conversion
// ---------------------------------------------------------------------------
__global__ void cvt_f32_bf16(const float* __restrict__ src,
                             unsigned short* __restrict__ dst, int n) {
    int i = blockIdx.x * blockDim.x + threadIdx.x;
    if (i < n) dst[i] = f2bfu(src[i]);
}

// ---------------------------------------------------------------------------
// K1: la[p,h] = log a = -softplus(wA[h].x[p,h] + bA[h])   (log-space decay)
// ---------------------------------------------------------------------------
__global__ void decay_kernel(const float* __restrict__ inp, const float* __restrict__ wA,
                             const float* __restrict__ bA, float* __restrict__ la_out) {
    int i = blockIdx.x * blockDim.x + threadIdx.x;   // i = p*H + h
    if (i >= TTOK * H_) return;
    int h = i & (H_ - 1);
    int p = i >> 4;
    const float* xr = inp + (size_t)p * D_ + h * CH_;
    const float* w  = wA + h * CH_;
    float acc = 0.f;
#pragma unroll
    for (int c = 0; c < CH_; c += 4) {
        float4 xv = *(const float4*)(xr + c);
        float4 wv = *(const float4*)(w + c);
        acc = fmaf(xv.x, wv.x, acc);
        acc = fmaf(xv.y, wv.y, acc);
        acc = fmaf(xv.z, wv.z, acc);
        acc = fmaf(xv.w, wv.w, acc);
    }
    acc += bA[h];
    float sp = (acc > 20.f) ? acc : log1pf(expf(acc));
    la_out[i] = -sp;
}

// ---------------------------------------------------------------------------
// K2: per-head projection GEMM via WMMA (16 tokens x 256 outputs per block).
// ---------------------------------------------------------------------------
__global__ __launch_bounds__(256) void proj_kernel(
    const float* __restrict__ inp,
    const unsigned short* __restrict__ wXBCb,   // (H,192,64) bf16
    const unsigned short* __restrict__ wGb,     // (H, 64,64) bf16
    const float* __restrict__ bXBC,             // (H,192)
    const float* __restrict__ bG,               // (H, 64)
    unsigned short* __restrict__ xbc_act,       // (TTOK,H,192) bf16 (post-SiLU)
    unsigned short* __restrict__ gate_act)      // (TTOK,H, 64) bf16
{
    const int h  = blockIdx.y;
    const int p0 = blockIdx.x * 16;
    const int wv = threadIdx.x >> 5;
    const int L  = threadIdx.x & 31;
    const int hf = L >> 4;
    const int ml = L & 15;

    const float* xrow = inp + (size_t)(p0 + ml) * D_ + h * CH_;
    v16bf a0 = packA_f32(xrow, hf * 8);
    v16bf a1 = packA_f32(xrow, 32 + hf * 8);

#pragma unroll
    for (int tt = 0; tt < 2; ++tt) {
        const int nt = wv * 2 + tt;            // 0..11 XBC, 12..15 gate
        const int n  = nt * 16 + ml;
        const unsigned short* wbase =
            (nt < 12) ? (wXBCb + ((size_t)h * 192 + n) * 64)
                      : (wGb   + ((size_t)h * 64 + (n - 192)) * 64);
        const int krun = hf * 8;
        v16bf b0 = loadB_bf16(wbase + krun);
        v16bf b1 = loadB_bf16(wbase + 32 + krun);

        v8f c = {};
        c = bfmma(a0, b0, c);
        c = bfmma(a1, b1, c);

#pragma unroll
        for (int r = 0; r < 8; ++r) {
            const int p = p0 + r + hf * 8;
            float v = c[r];
            if (nt < 12) {
                v += bXBC[h * 192 + n];
                v = v / (1.f + expf(-v));      // SiLU
                xbc_act[((size_t)p * H_ + h) * 192 + n] = f2bfu(v);
            } else {
                v += bG[h * 64 + (n - 192)];
                gate_act[((size_t)p * H_ + h) * 64 + (n - 192)] = f2bfu(v);
            }
        }
    }
}

// ---------------------------------------------------------------------------
// K3: chunked Mamba2 scan (Q=64) with WMMA. One block per (b,h) chain.
//   Per chunk:  G = C.B^T  -> M[i,j] = G * exp(lc[i]-lc[j]) (j<=i)
//               Y = M.X + (wcum*C).S^T        (gate fused on store)
//               S = exp(lc[63])*S + X^T.(wend*B)
// ---------------------------------------------------------------------------
__global__ __launch_bounds__(256) void scan_chunk_kernel(
    const unsigned short* __restrict__ xbc,    // (TTOK,H,192) bf16: X | B | C
    const unsigned short* __restrict__ gatev,  // (TTOK,H,64)  bf16
    const float* __restrict__ la_vals,         // (TTOK,H) log-decay
    unsigned short* __restrict__ yg)           // (TTOK,1024) bf16
{
    __shared__ unsigned short sXt  [Q_ * STR_];  // X^T  [c][t]
    __shared__ unsigned short sBrow[Q_ * STR_];  // B    [t][n]
    __shared__ unsigned short sBtd [Q_ * STR_];  // wend[t]*B^T [n][t]
    __shared__ unsigned short sCraw[Q_ * STR_];  // C    [t][n]
    __shared__ unsigned short sCd  [Q_ * STR_];  // wcum[t]*C [t][n]
    __shared__ unsigned short sM   [Q_ * STR_];  // M    [i][j]
    __shared__ float          sS   [Q_ * STR_];  // state [c][n] f32
    __shared__ float sla[Q_], slc[Q_], swcum[Q_], swend[Q_];

    const int bh  = blockIdx.x;                  // b*H + h
    const int b   = bh >> 4;
    const int h   = bh & 15;
    const int tid = threadIdx.x;
    const int wv  = tid >> 5;
    const int L   = tid & 31;
    const int hf  = L >> 4;
    const int ml  = L & 15;

    for (int e = tid; e < Q_ * STR_; e += 256) sS[e] = 0.f;
    __syncthreads();

    for (int ch = 0; ch < TSEQ / Q_; ++ch) {
        const int pbase = b * TSEQ + ch * Q_;

        // ---- decay prep: lc = cumsum(la), wcum = exp(lc), wend = exp(lc63-lc) ----
        if (tid < Q_) sla[tid] = la_vals[(size_t)(pbase + tid) * H_ + h];
        __syncthreads();
        if (tid == 0) {
            float run = 0.f;
            for (int t = 0; t < Q_; ++t) { run += sla[t]; slc[t] = run; }
        }
        __syncthreads();
        if (tid < Q_) {
            swcum[tid] = expf(slc[tid]);
            swend[tid] = expf(slc[Q_ - 1] - slc[tid]);
        }
        __syncthreads();

        // ---- stage chunk operands into LDS (scaled / transposed views) ----
        for (int e = tid; e < Q_ * 192; e += 256) {
            int t = e / 192, col = e - t * 192;
            unsigned short v = xbc[((size_t)(pbase + t) * H_ + h) * 192 + col];
            if (col < 64) {
                sXt[col * STR_ + t] = v;
            } else if (col < 128) {
                int n = col - 64;
                sBrow[t * STR_ + n] = v;
                sBtd[n * STR_ + t]  = f2bfu(bf2f(v) * swend[t]);
            } else {
                int n = col - 128;
                sCraw[t * STR_ + n] = v;
                sCd[t * STR_ + n]   = f2bfu(bf2f(v) * swcum[t]);
            }
        }
        __syncthreads();

        // ---- G = C.B^T, masked decay weights -> M (bf16 in LDS) ----
#pragma unroll
        for (int tt = 0; tt < 2; ++tt) {
            const int tile = wv * 2 + tt;
            const int mt = tile >> 2, nt = tile & 3;
            v8f g = {};
            g = bfmma(ldsA(sCraw, mt * 16 + ml, 0,  hf), ldsB(sBrow, nt * 16 + ml, 0,  hf), g);
            g = bfmma(ldsA(sCraw, mt * 16 + ml, 32, hf), ldsB(sBrow, nt * 16 + ml, 32, hf), g);
            const int j   = nt * 16 + ml;
            const float lj = slc[j];
#pragma unroll
            for (int r = 0; r < 8; ++r) {
                const int i = mt * 16 + r + 8 * hf;
                const float w = (j <= i) ? expf(slc[i] - lj) : 0.f;
                sM[i * STR_ + j] = f2bfu(g[r] * w);
            }
        }
        __syncthreads();

        // ---- Y = M.X + Cd.S^T  and  dS = X^T.Btd  (all WMMA) ----
        v8f yacc[2], sacc[2];
#pragma unroll
        for (int tt = 0; tt < 2; ++tt) {
            const int tile = wv * 2 + tt;
            const int mt = tile >> 2, nt = tile & 3;
            v8f acc = {};
            acc = bfmma(ldsA(sM,  mt * 16 + ml, 0,  hf), ldsB(sXt, nt * 16 + ml, 0,  hf), acc);
            acc = bfmma(ldsA(sM,  mt * 16 + ml, 32, hf), ldsB(sXt, nt * 16 + ml, 32, hf), acc);
            acc = bfmma(ldsA(sCd, mt * 16 + ml, 0,  hf), ldsB_f32(sS, nt * 16 + ml, 0,  hf), acc);
            acc = bfmma(ldsA(sCd, mt * 16 + ml, 32, hf), ldsB_f32(sS, nt * 16 + ml, 32, hf), acc);
            yacc[tt] = acc;

            v8f sa = {};
            sa = bfmma(ldsA(sXt, mt * 16 + ml, 0,  hf), ldsB(sBtd, nt * 16 + ml, 0,  hf), sa);
            sa = bfmma(ldsA(sXt, mt * 16 + ml, 32, hf), ldsB(sBtd, nt * 16 + ml, 32, hf), sa);
            sacc[tt] = sa;
        }

        // ---- epilogue: gate & store Y ----
#pragma unroll
        for (int tt = 0; tt < 2; ++tt) {
            const int tile = wv * 2 + tt;
            const int mt = tile >> 2, nt = tile & 3;
            const int c = nt * 16 + ml;
#pragma unroll
            for (int r = 0; r < 8; ++r) {
                const int p = pbase + mt * 16 + r + 8 * hf;
                float g = bf2f(gatev[((size_t)p * H_ + h) * 64 + c]);
                yg[(size_t)p * D_ + h * 64 + c] = f2bfu(yacc[tt][r] * g);
            }
        }
        __syncthreads();   // all readers of sS done

        // ---- S = exp(lc[63])*S + dS ----
        const float wfull = swcum[Q_ - 1];
#pragma unroll
        for (int tt = 0; tt < 2; ++tt) {
            const int tile = wv * 2 + tt;
            const int mt = tile >> 2, nt = tile & 3;
            const int n = nt * 16 + ml;
#pragma unroll
            for (int r = 0; r < 8; ++r) {
                const int c = mt * 16 + r + 8 * hf;
                sS[c * STR_ + n] = wfull * sS[c * STR_ + n] + sacc[tt][r];
            }
        }
        __syncthreads();
    }
}

// ---------------------------------------------------------------------------
// K4: out = Yg(bf16) @ Wout^T(bf16) + bout, f32 out.  M=16384, N=K=1024.
//   64x64 block tile, K staged 64-deep into LDS, double-buffered with
//   GLOBAL_LOAD_ASYNC_TO_LDS_B128 (ASYNCcnt) when available.
// ---------------------------------------------------------------------------
__global__ __launch_bounds__(256) void out_gemm(
    const unsigned short* __restrict__ yg,     // (TTOK,1024) bf16
    const unsigned short* __restrict__ woutb,  // (1024,1024) bf16, row = n, col = k
    const float* __restrict__ bout,
    float* __restrict__ out)                   // (TTOK,1024) f32
{
    __shared__ unsigned short sAb[2][64 * STR_];
    __shared__ unsigned short sBb[2][64 * STR_];

    const int tid = threadIdx.x;
    const int wv  = tid >> 5;
    const int L   = tid & 31;
    const int hf  = L >> 4;
    const int ml  = L & 15;

    const int mblk = blockIdx.x * 64;
    const int nblk = blockIdx.y * 64;
    const int mt = wv & 3;            // M sub-tile (16 rows)
    const int nh = wv >> 2;           // N half (32 cols -> 2 WMMA tiles)

    // stage one 64(A-rows)+64(B-rows) x 64-deep K panel into LDS buffer `buf`
    auto stage = [&](int buf, int kb) {
#pragma unroll
        for (int i = 0; i < 2; ++i) {
            const int idx = tid + i * 256;          // 512 chunks of 8 bf16
            const int row = idx >> 3;
            const int kk  = (idx & 7) * 8;
            async_cp_b128(yg    + (size_t)(mblk + row) * D_ + kb + kk,
                          &sAb[buf][row * STR_ + kk]);
            async_cp_b128(woutb + (size_t)(nblk + row) * D_ + kb + kk,
                          &sBb[buf][row * STR_ + kk]);
        }
    };

    v8f c0 = {};
    v8f c1 = {};

    stage(0, 0);
    wait_async_all();
    __syncthreads();

    for (int it = 0; it < D_ / 64; ++it) {
        const int cur = it & 1;
        if (it + 1 < D_ / 64) stage(cur ^ 1, (it + 1) * 64);   // buffer cur^1 was
                                                               // released at it-1's barrier
#pragma unroll
        for (int kb2 = 0; kb2 < 64; kb2 += 32) {
            v16bf a  = ldsA(sAb[cur], mt * 16 + ml, kb2, hf);
            v16bf b0 = ldsB(sBb[cur], nh * 32 + ml,      kb2, hf);
            v16bf b1 = ldsB(sBb[cur], nh * 32 + 16 + ml, kb2, hf);
            c0 = bfmma(a, b0, c0);
            c1 = bfmma(a, b1, c1);
        }
        if (it + 1 < D_ / 64) wait_async_all();
        __syncthreads();
    }

#pragma unroll
    for (int r = 0; r < 8; ++r) {
        const int m = mblk + mt * 16 + r + 8 * hf;
        const int n = nblk + nh * 32 + ml;
        out[(size_t)m * D_ + n]      = c0[r] + bout[n];
        out[(size_t)m * D_ + n + 16] = c1[r] + bout[n + 16];
    }
}

// ---------------------------------------------------------------------------
extern "C" void kernel_launch(void* const* d_in, const int* in_sizes, int n_in,
                              void* d_out, int out_size, void* d_ws, size_t ws_size,
                              hipStream_t stream) {
    const float* inp  = (const float*)d_in[0];
    const float* wA   = (const float*)d_in[1];
    const float* bA   = (const float*)d_in[2];
    const float* wXBC = (const float*)d_in[3];
    const float* bXBC = (const float*)d_in[4];
    const float* wG   = (const float*)d_in[5];
    const float* bG   = (const float*)d_in[6];
    const float* Wout = (const float*)d_in[7];
    const float* bout = (const float*)d_in[8];

    char* ws = (char*)d_ws;
    size_t off = 0;
    auto alloc = [&](size_t bytes) -> void* {
        void* p = ws + off;
        off += (bytes + 255) & ~(size_t)255;
        return p;
    };
    unsigned short* wXBCb = (unsigned short*)alloc((size_t)H_ * 192 * 64 * 2);
    unsigned short* wGb   = (unsigned short*)alloc((size_t)H_ * 64 * 64 * 2);
    unsigned short* Woutb = (unsigned short*)alloc((size_t)D_ * D_ * 2);
    unsigned short* xbc   = (unsigned short*)alloc((size_t)TTOK * H_ * 192 * 2);
    unsigned short* gatev = (unsigned short*)alloc((size_t)TTOK * H_ * 64 * 2);
    float*          lav   = (float*)alloc((size_t)TTOK * H_ * 4);
    unsigned short* ygb   = (unsigned short*)alloc((size_t)TTOK * D_ * 2);

    cvt_f32_bf16<<<(H_ * 192 * 64 + 255) / 256, 256, 0, stream>>>(wXBC, wXBCb, H_ * 192 * 64);
    cvt_f32_bf16<<<(H_ * 64 * 64 + 255) / 256, 256, 0, stream>>>(wG, wGb, H_ * 64 * 64);
    cvt_f32_bf16<<<(D_ * D_ + 255) / 256, 256, 0, stream>>>(Wout, Woutb, D_ * D_);

    decay_kernel<<<(TTOK * H_) / 256, 256, 0, stream>>>(inp, wA, bA, lav);
    proj_kernel<<<dim3(TTOK / 16, H_), 256, 0, stream>>>(inp, wXBCb, wGb, bXBC, bG, xbc, gatev);
    scan_chunk_kernel<<<64, 256, 0, stream>>>(xbc, gatev, lav, ygb);
    out_gemm<<<dim3(TTOK / 64, D_ / 64), 256, 0, stream>>>(ygb, Woutb, bout, (float*)d_out);
}
